// MyDecoderLayer_81930796138810
// MI455X (gfx1250) — matrix operand order repaired and checked by reading, compiled, not run on hardware
//
#include <hip/hip_runtime.h>
#include <hip/hip_bf16.h>
#include <math.h>

#ifndef __has_builtin
#define __has_builtin(x) 0
#endif
#if __has_builtin(__builtin_amdgcn_global_load_async_to_lds_b128)
#define USE_ASYNC_LDS 1
#endif

// ---------------------------------------------------------------------------
// Types for CDNA5 WMMA (wave32): A/B = v16h (16 x f16), C/D = v8f (8 x f32)
// ---------------------------------------------------------------------------
typedef __attribute__((ext_vector_type(16))) _Float16 v16h;
typedef __attribute__((ext_vector_type(8)))  _Float16 v8h;
typedef __attribute__((ext_vector_type(8)))  float    v8f;

#define HH 32
#define WW 32
#define NTOK 1024
#define BB 16
#define DD 384
#define C2 768
#define C4 1536
#define NHEADS 8
#define DHEAD 48
#define ROWS (BB * NTOK)   // 16384

#ifdef USE_ASYNC_LDS
// Param type per clang diagnostic: pointer to int vector_size(16), global /
// shared address spaces ("__device__" / "__shared__" in HIP diag spelling).
typedef int v4i_vs __attribute__((vector_size(16)));
typedef __attribute__((address_space(1))) v4i_vs* as1_v4i_p;
typedef __attribute__((address_space(3))) v4i_vs* as3_v4i_p;
__device__ __forceinline__ void async_ld_b128(const _Float16* g, _Float16* l) {
    __builtin_amdgcn_global_load_async_to_lds_b128(
        (as1_v4i_p)(void*)(const_cast<_Float16*>(g)),
        (as3_v4i_p)(void*)l, 0, 0);
}
__device__ __forceinline__ void wait_async0() {
#if __has_builtin(__builtin_amdgcn_s_wait_asynccnt)
    __builtin_amdgcn_s_wait_asynccnt(0);
#else
    asm volatile("s_wait_asynccnt 0" ::: "memory");
#endif
}
#endif

// ---------------------------------------------------------------------------
// Block-wide sum reduction (blockDim.x == 256, wave32)
// ---------------------------------------------------------------------------
__device__ __forceinline__ float block_reduce_sum(float v, float* sh) {
    #pragma unroll
    for (int o = 16; o; o >>= 1) v += __shfl_xor(v, o);
    int lane = threadIdx.x & 31, wid = threadIdx.x >> 5;
    if (lane == 0) sh[wid] = v;
    __syncthreads();
    float r = 0.f;
    if (threadIdx.x < 8) r = sh[threadIdx.x];
    #pragma unroll
    for (int o = 4; o; o >>= 1) r += __shfl_xor(r, o);
    if (threadIdx.x == 0) sh[0] = r;
    __syncthreads();
    r = sh[0];
    __syncthreads();
    return r;
}

__device__ __forceinline__ v16h join16(const _Float16* p0, const _Float16* p1) {
    v8h a = *(const v8h*)p0;
    v8h b = *(const v8h*)p1;
    v16h r;
    #pragma unroll
    for (int j = 0; j < 8; ++j) { r[j] = a[j]; r[8 + j] = b[j]; }
    return r;
}

// ---------------------------------------------------------------------------
// Generic batched WMMA GEMM.  C[bz](MxN) = A[bz](MxK f16) * B[bz] (+bias,+res)
// BT=true  -> B is [N][K] row-major (contiguous staging; all weights + ctxT).
// BT=false -> B is [K][N] row-major (transpose-staged; context GEMM only).
// OUT_SWAP -> element (row,col) stored at Cout[col*M+row] (f16 v8h stores).
// Block: 256 threads = 8 waves (4 M-waves x 2 N-waves). Tile 128x128, K=64
// per stage, 16 wmma per stage. Global->LDS is double-buffered: async DMA
// (GLOBAL_LOAD_ASYNC_TO_LDS_B128 + s_wait_asynccnt) when available, else
// register-staged prefetch. Requires M%128==0, N%128==0, K%64==0.
// ---------------------------------------------------------------------------
template<bool HAS_BIAS, bool HAS_RES, bool OUT_F16, bool BT, bool OUT_SWAP>
__global__ __launch_bounds__(256) void gemm_wmma(
    const _Float16* __restrict__ A, const _Float16* __restrict__ Bm,
    void* __restrict__ Cout, const float* __restrict__ bias,
    const float* __restrict__ res,
    int M, int N, int K,
    long long strideA, long long strideB, long long strideC)
{
#ifdef USE_ASYNC_LDS
    __shared__ _Float16 Alds[2][128][72];
    __shared__ _Float16 Blds[2][128][72];
#else
    __shared__ _Float16 Alds[1][128][72];
    __shared__ _Float16 Blds[1][128][72];
#endif

    const int tid  = threadIdx.x;
    const int lane = tid & 31, wid = tid >> 5;
    const int row_l = lane & 15;
    const int hi    = lane >> 4;           // 0/1
    const int waveM = wid >> 1;            // 0..3
    const int waveN = wid & 1;             // 0..1
    const int m0 = blockIdx.y * 128;
    const int n0 = blockIdx.x * 128;
    const int bz = blockIdx.z;

    const _Float16* Ab = A  + (long long)bz * strideA;
    const _Float16* Bb = Bm + (long long)bz * strideB;

    v8f acc[2][4];
    #pragma unroll
    for (int mt = 0; mt < 2; ++mt)
        #pragma unroll
        for (int nt = 0; nt < 4; ++nt)
            #pragma unroll
            for (int i = 0; i < 8; ++i) acc[mt][nt][i] = 0.f;

    const int st_r = tid >> 1;             // staging row 0..127
    const int st_c = (tid & 1) * 32;       // staging col 0 or 32

    // 16 WMMAs on the current LDS buffer
    auto compute = [&](int cb) {
        #pragma unroll
        for (int kc = 0; kc < 2; ++kc) {
            v16h afrag[2];
            #pragma unroll
            for (int mt = 0; mt < 2; ++mt) {
                int r  = waveM * 32 + mt * 16 + row_l;
                int kb = kc * 32 + hi * 8;
                afrag[mt] = join16(&Alds[cb][r][kb], &Alds[cb][r][kb + 16]);
            }
            #pragma unroll
            for (int nt = 0; nt < 4; ++nt) {
                int c  = waveN * 64 + nt * 16 + row_l;
                int kb = kc * 32 + hi * 8;
                v16h bfrag = join16(&Blds[cb][c][kb], &Blds[cb][c][kb + 16]);
                #pragma unroll
                for (int mt = 0; mt < 2; ++mt)
                    acc[mt][nt] = __builtin_amdgcn_wmma_f32_16x16x32_f16(
                        false, afrag[mt], false, bfrag,
                        (short)0, acc[mt][nt], false, false);
            }
        }
    };

    const int ntile = K >> 6;

#ifdef USE_ASYNC_LDS
    if constexpr (BT) {
        // ---- async DMA double-buffered pipeline: 1 barrier / stage ----
        auto issue = [&](int k0, int buf) {
            const _Float16* ap = Ab + (long long)(m0 + st_r) * K + k0 + st_c;
            const _Float16* bp = Bb + (long long)(n0 + st_r) * K + k0 + st_c;
            #pragma unroll
            for (int j = 0; j < 4; ++j) {
                async_ld_b128(ap + 8 * j, &Alds[buf][st_r][st_c + 8 * j]);
                async_ld_b128(bp + 8 * j, &Blds[buf][st_r][st_c + 8 * j]);
            }
        };
        issue(0, 0);
        for (int it = 0; it < ntile; ++it) {
            wait_async0();           // this wave's DMA for buf[it&1] landed
            __syncthreads();         // all waves' DMA landed; prev compute done
            if (it + 1 < ntile) issue((it + 1) << 6, (it + 1) & 1);
            compute(it & 1);         // overlaps with next tile's DMA
        }
    } else
#endif
    if constexpr (BT) {
        // ---- register-staged double buffering (fallback) ----
        v8h areg[4], breg[4];
        auto fetch = [&](int k0) {
            const _Float16* ap = Ab + (long long)(m0 + st_r) * K + k0 + st_c;
            const _Float16* bp = Bb + (long long)(n0 + st_r) * K + k0 + st_c;
            #pragma unroll
            for (int j = 0; j < 4; ++j) {
                areg[j] = *(const v8h*)(ap + 8 * j);
                breg[j] = *(const v8h*)(bp + 8 * j);
            }
            if (k0 + 64 < K) {
                __builtin_prefetch(ap + 64, 0, 3);
                __builtin_prefetch(bp + 64, 0, 3);
            }
        };
        fetch(0);
        for (int it = 0; it < ntile; ++it) {
            __syncthreads();         // prev compute done: safe to overwrite LDS
            #pragma unroll
            for (int j = 0; j < 4; ++j) {
                *(v8h*)&Alds[0][st_r][st_c + 8 * j] = areg[j];
                *(v8h*)&Blds[0][st_r][st_c + 8 * j] = breg[j];
            }
            __syncthreads();
            if (it + 1 < ntile) fetch((it + 1) << 6);  // overlaps compute
            compute(0);
        }
    } else {
        // ---- B in [K][N]: transpose-staged (context GEMM only) ----
        for (int k0 = 0; k0 < K; k0 += 64) {
            __syncthreads();
            {
                const _Float16* ap = Ab + (long long)(m0 + st_r) * K + k0 + st_c;
                #pragma unroll
                for (int j = 0; j < 4; ++j)
                    *(v8h*)&Alds[0][st_r][st_c + 8 * j] = *(const v8h*)(ap + 8 * j);
                if (k0 + 64 < K) __builtin_prefetch(ap + 64, 0, 3);
            }
            #pragma unroll
            for (int g = 0; g < 4; ++g) {
                int gg = tid + g * 256;              // 0..1023
                int kk = gg >> 4;                    // 0..63
                int nn = (gg & 15) * 8;              // 0..120
                const _Float16* bp = Bb + (long long)(k0 + kk) * N + n0 + nn;
                v8h b8 = *(const v8h*)bp;
                if (g == 0 && k0 + 64 < K)
                    __builtin_prefetch(bp + (long long)64 * N, 0, 3);
                #pragma unroll
                for (int j = 0; j < 8; ++j) Blds[0][nn + j][kk] = b8[j];
            }
            __syncthreads();
            compute(0);
        }
    }

    // --- epilogue ---
    #pragma unroll
    for (int mt = 0; mt < 2; ++mt)
        #pragma unroll
        for (int nt = 0; nt < 4; ++nt) {
            int col = n0 + waveN * 64 + nt * 16 + row_l;
            float bv = 0.f;
            if constexpr (HAS_BIAS) bv = bias[col];
            if constexpr (OUT_SWAP) {
                int rowbase = m0 + waveM * 32 + mt * 16 + hi * 8;
                v8h o;
                #pragma unroll
                for (int i = 0; i < 8; ++i)
                    o[i] = (_Float16)(acc[mt][nt][i] + bv);
                *(v8h*)&((_Float16*)Cout)[(long long)bz * strideC +
                                          (long long)col * M + rowbase] = o;
            } else {
                #pragma unroll
                for (int i = 0; i < 8; ++i) {
                    int row = m0 + waveM * 32 + mt * 16 + i + hi * 8;
                    long long cidx = (long long)bz * strideC +
                                     (long long)row * N + col;
                    float v = acc[mt][nt][i] + bv;
                    if constexpr (HAS_RES) v += res[cidx];
                    if constexpr (OUT_F16) ((_Float16*)Cout)[cidx] = (_Float16)v;
                    else                   ((float*)Cout)[cidx]    = v;
                }
            }
        }
}

// ---------------------------------------------------------------------------
// LayerNorm over last dim (C = 384 or 768), one block per row, f16/f32 out.
// ---------------------------------------------------------------------------
template<bool W16, bool W32>
__global__ __launch_bounds__(256) void ln_kernel(
    const float* __restrict__ x, const float* __restrict__ g,
    const float* __restrict__ bta, _Float16* __restrict__ o16,
    float* __restrict__ o32, int C)
{
    __shared__ float sh[8];
    long long row = blockIdx.x;
    const float* xr = x + row * C;
    float s = 0.f, s2 = 0.f;
    for (int j = threadIdx.x; j < C; j += 256) {
        float v = xr[j]; s += v; s2 += v * v;
    }
    s  = block_reduce_sum(s, sh);
    s2 = block_reduce_sum(s2, sh);
    float mu   = s / C;
    float var  = s2 / C - mu * mu;
    float rstd = rsqrtf(var + 1e-5f);
    for (int j = threadIdx.x; j < C; j += 256) {
        float y = (xr[j] - mu) * rstd * g[j] + bta[j];
        if constexpr (W16) o16[row * C + j] = (_Float16)y;
        if constexpr (W32) o32[row * C + j] = y;
    }
}

// ---------------------------------------------------------------------------
// Cross-attention epilogue fused: tx = concat(ch1,ch2) + LN_cn(rep)
// ---------------------------------------------------------------------------
__global__ __launch_bounds__(256) void tx_fuse(
    const float* __restrict__ rep, const float* __restrict__ g,
    const float* __restrict__ bta, const float* __restrict__ ch1,
    const float* __restrict__ ch2, float* __restrict__ tx)
{
    __shared__ float sh[8];
    long long row = blockIdx.x;
    const float* xr = rep + row * C2;
    float s = 0.f, s2 = 0.f;
    for (int j = threadIdx.x; j < C2; j += 256) {
        float v = xr[j]; s += v; s2 += v * v;
    }
    s  = block_reduce_sum(s, sh);
    s2 = block_reduce_sum(s2, sh);
    float mu   = s / C2;
    float rstd = rsqrtf(s2 / C2 - mu * mu + 1e-5f);
    for (int j = threadIdx.x; j < C2; j += 256) {
        float y = (xr[j] - mu) * rstd * g[j] + bta[j];
        float base = (j < DD) ? ch1[row * DD + j] : ch2[row * DD + (j - DD)];
        tx[row * C2 + j] = base + y;
    }
}

// ---------------------------------------------------------------------------
// Cross-attention softmaxes.
// ---------------------------------------------------------------------------
__global__ __launch_bounds__(256) void key_softmax(
    const float* __restrict__ n2, _Float16* __restrict__ key16)
{
    int col  = blockIdx.x * 8 + (threadIdx.x >> 5);  // 0 .. B*D-1
    int lane = threadIdx.x & 31;
    int b = col / DD, d = col % DD;
    const float* src = n2 + ((long long)b * NTOK) * DD + d;
    float mx = -1e30f;
    for (int n = lane; n < NTOK; n += 32) mx = fmaxf(mx, src[(long long)n * DD]);
    #pragma unroll
    for (int o = 16; o; o >>= 1) mx = fmaxf(mx, __shfl_xor(mx, o));
    float s = 0.f;
    for (int n = lane; n < NTOK; n += 32) s += expf(src[(long long)n * DD] - mx);
    #pragma unroll
    for (int o = 16; o; o >>= 1) s += __shfl_xor(s, o);
    float inv = 1.f / s;
    _Float16* dst = key16 + ((long long)b * DD + d) * NTOK;
    for (int n = lane; n < NTOK; n += 32)
        dst[n] = (_Float16)(expf(src[(long long)n * DD] - mx) * inv);
}

__global__ __launch_bounds__(256) void query_softmax(
    const float* __restrict__ n2, _Float16* __restrict__ q16)
{
    long long row = (long long)blockIdx.x * 8 + (threadIdx.x >> 5);  // 0..16383
    int lane = threadIdx.x & 31;
    const float* src = n2 + row * DD;
    float mx = -1e30f;
    for (int j = lane; j < DD; j += 32) mx = fmaxf(mx, src[j]);
    #pragma unroll
    for (int o = 16; o; o >>= 1) mx = fmaxf(mx, __shfl_xor(mx, o));
    float s = 0.f;
    for (int j = lane; j < DD; j += 32) s += expf(src[j] - mx);
    #pragma unroll
    for (int o = 16; o; o >>= 1) s += __shfl_xor(s, o);
    float inv = 1.f / s;
    _Float16* dst = q16 + row * DD;
    for (int j = lane; j < DD; j += 32)
        dst[j] = (_Float16)(expf(src[j] - mx) * inv);
}

// ---------------------------------------------------------------------------
// Channel attention: one block per (b, head). qkv layout [B, N, 1152].
// ---------------------------------------------------------------------------
__global__ __launch_bounds__(256) void chan_attn(
    const float* __restrict__ qkv, const float* __restrict__ temp,
    _Float16* __restrict__ xatt16)
{
    __shared__ float    rqs64[64], rks64[64];  // [48..63] = 0
    __shared__ _Float16 alds[64][72];     // qn^T tile (c x 64n), rows>=48 zero
    __shared__ _Float16 blds[64][72];     // kn^T tile (e x 64n)
    __shared__ float    attl[64][72];     // logits (only [0..47][0..47] used)
    __shared__ _Float16 att16[64][72];    // softmaxed attn, zero-padded 64x64

    const int bh = blockIdx.x;
    const int b  = bh >> 3, h = bh & 7;
    const float* base = qkv + (long long)b * NTOK * (3 * DD);
    const int qoff = h * DHEAD;
    const int koff = DD + h * DHEAD;
    const int voff = 2 * DD + h * DHEAD;

    const int tid = threadIdx.x, lane = tid & 31, wid = tid >> 5;
    const int row_l = lane & 15, hi = lane >> 4;

    for (int i = tid; i < 64 * 72; i += 256) (&att16[0][0])[i] = (_Float16)0.f;
    if (tid >= 48 && tid < 64) { rqs64[tid] = 0.f; rks64[tid] = 0.f; }

    // --- L2 norms along tokens: 96 columns (48 q + 48 k), one wave each ---
    for (int col = wid; col < 96; col += 8) {
        int off = (col < DHEAD) ? (qoff + col) : (koff + col - DHEAD);
        float s = 0.f;
        for (int n = lane; n < NTOK; n += 32) {
            float v = base[(long long)n * (3 * DD) + off];
            s += v * v;
        }
        #pragma unroll
        for (int o = 16; o; o >>= 1) s += __shfl_xor(s, o);
        if (lane == 0) {
            float sc = 1.f / fmaxf(sqrtf(s), 1e-12f);
            if (col < DHEAD) rqs64[col] = sc; else rks64[col - DHEAD] = sc;
        }
    }
    __syncthreads();

    // --- GEMM1: logits[c,e] = sum_n qn[c,n]*kn[e,n], M=N=48(pad64), K=1024 ---
    const int mtile = wid >> 1;        // 0..3
    const int wN    = wid & 1;         // 0..1 (2 ntiles each)
    v8f acc2[2];
    #pragma unroll
    for (int j = 0; j < 2; ++j)
        #pragma unroll
        for (int i = 0; i < 8; ++i) acc2[j][i] = 0.f;

    for (int n0 = 0; n0 < NTOK; n0 += 64) {   // 16 stages of 64 tokens
        __syncthreads();
        #pragma unroll
        for (int pp = 0; pp < 2; ++pp) {
            int pr = tid + pp * 256;          // 0..511
            int kk = pr >> 3;                 // 0..63
            int cg = (pr & 7) * 8;            // 0,8,...,56
            int ca = (cg < DHEAD) ? cg : 40;  // clamped, always in-bounds
            long long nidx = (long long)(n0 + kk) * (3 * DD);
            const float4* qp = (const float4*)(base + nidx + qoff + ca);
            const float4* kp = (const float4*)(base + nidx + koff + ca);
            float4 q0 = qp[0], q1 = qp[1];
            float4 k0 = kp[0], k1 = kp[1];
            float qa[8] = {q0.x, q0.y, q0.z, q0.w, q1.x, q1.y, q1.z, q1.w};
            float ka[8] = {k0.x, k0.y, k0.z, k0.w, k1.x, k1.y, k1.z, k1.w};
            #pragma unroll
            for (int j = 0; j < 8; ++j) {
                int c = cg + j;               // pad cols scaled by 0
                alds[c][kk] = (_Float16)(qa[j] * rqs64[c]);
                blds[c][kk] = (_Float16)(ka[j] * rks64[c]);
            }
        }
        __syncthreads();

        #pragma unroll
        for (int kc = 0; kc < 2; ++kc) {
            int ar = mtile * 16 + row_l, kb = kc * 32 + hi * 8;
            v16h af = join16(&alds[ar][kb], &alds[ar][kb + 16]);
            #pragma unroll
            for (int j = 0; j < 2; ++j) {
                int nt = wN * 2 + j;
                int bc = nt * 16 + row_l, kbb = kc * 32 + hi * 8;
                v16h bfv = join16(&blds[bc][kbb], &blds[bc][kbb + 16]);
                acc2[j] = __builtin_amdgcn_wmma_f32_16x16x32_f16(
                    false, af, false, bfv, (short)0, acc2[j], false, false);
            }
        }
    }
    __syncthreads();

    float tempv = temp[h];
    #pragma unroll
    for (int j = 0; j < 2; ++j) {
        int nt = wN * 2 + j;
        int col = nt * 16 + row_l;
        #pragma unroll
        for (int i = 0; i < 8; ++i) {
            int row = mtile * 16 + i + hi * 8;
            attl[row][col] = acc2[j][i] * tempv;  // unconditional (padded LDS)
        }
    }
    __syncthreads();

    // --- softmax over e (48) per row ---
    if (tid < DHEAD) {
        float mx = -1e30f;
        for (int e = 0; e < DHEAD; ++e) mx = fmaxf(mx, attl[tid][e]);
        float s = 0.f;
        for (int e = 0; e < DHEAD; ++e) s += expf(attl[tid][e] - mx);
        float inv = 1.f / s;
        for (int e = 0; e < DHEAD; ++e)
            att16[tid][e] = (_Float16)(expf(attl[tid][e] - mx) * inv);
    }
    __syncthreads();

    // --- GEMM2: out[c,n] = sum_e attn[c,e]*v[e,n], M=48, K=48(pad64), N=1024
    v16h afr[3][2];
    #pragma unroll
    for (int mt = 0; mt < 3; ++mt)
        #pragma unroll
        for (int kc = 0; kc < 2; ++kc) {
            int r  = mt * 16 + row_l;
            int kb = kc * 32 + hi * 16;
            afr[mt][kc] = join16(&att16[r][kb], &att16[r][kb + 8]);
        }

    for (int ct = wid; ct < 64; ct += 8) {
        int n = ct * 16 + row_l;
        const float* vptr = base + (long long)n * (3 * DD) + voff;
        v8f oacc[3];
        #pragma unroll
        for (int mt = 0; mt < 3; ++mt)
            #pragma unroll
            for (int i = 0; i < 8; ++i) oacc[mt][i] = 0.f;

        #pragma unroll
        for (int kc = 0; kc < 2; ++kc) {
            int ebase = kc * 32 + hi * 16;
            int eb    = (ebase < DHEAD) ? ebase : 0;       // clamped address
            bool valid = (ebase < DHEAD);
            const float4* vp4 = (const float4*)(vptr + eb);
            float4 f0 = vp4[0], f1 = vp4[1], f2 = vp4[2], f3 = vp4[3];
            float tmpv[16] = {f0.x, f0.y, f0.z, f0.w, f1.x, f1.y, f1.z, f1.w,
                              f2.x, f2.y, f2.z, f2.w, f3.x, f3.y, f3.z, f3.w};
            v16h bf;
            #pragma unroll
            for (int j = 0; j < 16; ++j)
                bf[j] = valid ? (_Float16)tmpv[j] : (_Float16)0.f;
            #pragma unroll
            for (int mt = 0; mt < 3; ++mt)
                oacc[mt] = __builtin_amdgcn_wmma_f32_16x16x32_f16(
                    false, afr[mt][kc], false, bf, (short)0, oacc[mt],
                    false, false);
        }
        _Float16* orow = xatt16 + ((long long)b * NTOK + n) * DD + h * DHEAD;
        #pragma unroll
        for (int mt = 0; mt < 3; ++mt)
            #pragma unroll
            for (int i = 0; i < 8; ++i) {
                int c = mt * 16 + i + hi * 8;   // always < 48
                orow[c] = (_Float16)oacc[mt][i];
            }
    }
}

// ---------------------------------------------------------------------------
// Depthwise 3x3 SAME conv on [B, C4, 32, 32] (token-major) + exact GELU
// ---------------------------------------------------------------------------
__global__ __launch_bounds__(256) void dwconv_gelu(
    const float* __restrict__ h1, const float* __restrict__ w,
    const float* __restrict__ bias, _Float16* __restrict__ out)
{
    long long idx = (long long)blockIdx.x * 256 + threadIdx.x;
    int c = (int)(idx % C4);
    long long rest = idx / C4;
    int n  = (int)(rest & (NTOK - 1));
    int b  = (int)(rest >> 10);
    int hh = n >> 5, ww = n & 31;
    float acc = bias[c];
    #pragma unroll
    for (int dy = -1; dy <= 1; ++dy)
        #pragma unroll
        for (int dx = -1; dx <= 1; ++dx) {
            int y = hh + dy, x = ww + dx;
            if (y >= 0 && y < HH && x >= 0 && x < WW)
                acc += h1[((long long)b * NTOK + y * WW + x) * C4 + c] *
                       w[c * 9 + (dy + 1) * 3 + (dx + 1)];
        }
    float g = 0.5f * acc * (1.f + erff(acc * 0.70710678118654752f));
    out[idx] = (_Float16)g;
}

// ---------------------------------------------------------------------------
// Weight conversions
// ---------------------------------------------------------------------------
__global__ void f32_to_f16(const float* __restrict__ s, _Float16* __restrict__ d,
                           long long n) {
    long long i = (long long)blockIdx.x * 256 + threadIdx.x;
    if (i < n) d[i] = (_Float16)s[i];
}
// dst[Cc x R] = transpose of src[R x Cc], f32 -> f16
__global__ void transpose_f16(const float* __restrict__ s, _Float16* __restrict__ d,
                              int R, int Cc) {
    long long i = (long long)blockIdx.x * 256 + threadIdx.x;
    if (i < (long long)R * Cc) {
        int r = (int)(i / Cc), c = (int)(i % Cc);
        d[(long long)c * R + r] = (_Float16)s[i];
    }
}

// ---------------------------------------------------------------------------
// Host-side orchestration
// ---------------------------------------------------------------------------
extern "C" void kernel_launch(void* const* d_in, const int* in_sizes, int n_in,
                              void* d_out, int out_size, void* d_ws, size_t ws_size,
                              hipStream_t stream) {
    const float* x1     = (const float*)d_in[0];
    const float* x2     = (const float*)d_in[1];
    const float* ln1_g  = (const float*)d_in[2];
    const float* ln1_b  = (const float*)d_in[3];
    const float* qkv_w  = (const float*)d_in[4];
    const float* temp   = (const float*)d_in[5];
    const float* proj_w = (const float*)d_in[6];
    const float* proj_b = (const float*)d_in[7];
    const float* ln3_g  = (const float*)d_in[8];
    const float* ln3_b  = (const float*)d_in[9];
    const float* rp_w   = (const float*)d_in[10];
    const float* rp_b   = (const float*)d_in[11];
    const float* cn_g   = (const float*)d_in[12];
    const float* cn_b   = (const float*)d_in[13];
    const float* ln2_g  = (const float*)d_in[14];
    const float* ln2_b  = (const float*)d_in[15];
    const float* fc1_w  = (const float*)d_in[16];
    const float* fc1_b  = (const float*)d_in[17];
    const float* dw_w   = (const float*)d_in[18];
    const float* dw_b   = (const float*)d_in[19];
    const float* fc2_w  = (const float*)d_in[20];
    const float* fc2_b  = (const float*)d_in[21];

    // ---- carve workspace ----
    char* p = (char*)d_ws;
    auto alloc = [&](size_t bytes) -> char* {
        char* r = p; p += (bytes + 255) & ~(size_t)255; return r;
    };
    _Float16* w16_qkvT  = (_Float16*)alloc((size_t)DD * 3 * DD * 2); // [1152][384]
    _Float16* w16_projT = (_Float16*)alloc((size_t)DD * DD * 2);     // [384][384]
    _Float16* w16_rp    = (_Float16*)alloc((size_t)C2 * DD * 2);     // [768][384]
    _Float16* w16_fc1T  = (_Float16*)alloc((size_t)C4 * C2 * 2);     // [1536][768]
    _Float16* w16_fc2T  = (_Float16*)alloc((size_t)C2 * C4 * 2);     // [768][1536]
    _Float16* a16       = (_Float16*)alloc((size_t)ROWS * C4 * 2);   // f16 staging
    float*    big       = (float*)   alloc((size_t)ROWS * C4 * 4);   // qkv then h1
    float*    qkvbuf    = big;                                       // [ROWS,1152]
    float*    h1        = big;                                       // [ROWS,1536]
    _Float16* xatt16    = (_Float16*)alloc((size_t)ROWS * DD * 2);
    float*    ch1       = (float*)   alloc((size_t)ROWS * DD * 4);
    float*    ch2       = (float*)   alloc((size_t)ROWS * DD * 4);
    _Float16* n1_16     = (_Float16*)alloc((size_t)ROWS * DD * 2);
    float*    n2f       = (float*)   alloc((size_t)ROWS * DD * 4);
    _Float16* key16     = (_Float16*)alloc((size_t)BB * DD * NTOK * 2);
    _Float16* q16       = (_Float16*)alloc((size_t)ROWS * DD * 2);
    _Float16* ctxT16    = (_Float16*)alloc((size_t)BB * DD * DD * 2); // [e][d]
    _Float16* attd16    = (_Float16*)alloc((size_t)ROWS * DD * 2);
    float*    rep       = (float*)   alloc((size_t)ROWS * C2 * 4);
    float*    tx        = (float*)   alloc((size_t)ROWS * C2 * 4);
    (void)ws_size; (void)n_in; (void)in_sizes; (void)out_size;

    // ---- weight conversion (BT layouts) ----
    auto tpose = [&](const float* s, _Float16* d, int R, int Cc) {
        transpose_f16<<<(unsigned)(((long long)R * Cc + 255) / 256), 256, 0, stream>>>(
            s, d, R, Cc);
    };
    tpose(qkv_w, w16_qkvT, DD, 3 * DD);   // [384,1152] -> [1152][384]
    tpose(proj_w, w16_projT, DD, DD);     // [384,384]  -> [384][384]^T
    tpose(fc1_w, w16_fc1T, C2, C4);       // [768,1536] -> [1536][768]
    tpose(fc2_w, w16_fc2T, C4, C2);       // [1536,768] -> [768][1536]
    f32_to_f16<<<(unsigned)(((long long)C2 * DD + 255) / 256), 256, 0, stream>>>(
        rp_w, w16_rp, (long long)C2 * DD);  // rp_w is already [N=768][K=384]

    // ---- per-stream: LN1 -> QKV -> channel attention -> proj(+residual) ----
    for (int s = 0; s < 2; ++s) {
        const float* xin = s ? x2 : x1;
        float* chout     = s ? ch2 : ch1;
        ln_kernel<true, false><<<ROWS, 256, 0, stream>>>(
            xin, ln1_g, ln1_b, a16, nullptr, DD);
        gemm_wmma<false, false, false, true, false>
            <<<dim3(3 * DD / 128, ROWS / 128, 1), 256, 0, stream>>>(
            a16, w16_qkvT, qkvbuf, nullptr, nullptr, ROWS, 3 * DD, DD, 0, 0, 0);
        chan_attn<<<BB * NHEADS, 256, 0, stream>>>(qkvbuf, temp, xatt16);
        gemm_wmma<true, true, false, true, false>
            <<<dim3(DD / 128, ROWS / 128, 1), 256, 0, stream>>>(
            xatt16, w16_projT, chout, proj_b, xin, ROWS, DD, DD, 0, 0, 0);
    }

    // ---- cross attention ----
    ln_kernel<true, false><<<ROWS, 256, 0, stream>>>(ch1, ln3_g, ln3_b, n1_16, nullptr, DD);
    ln_kernel<false, true><<<ROWS, 256, 0, stream>>>(ch2, ln3_g, ln3_b, nullptr, n2f, DD);
    key_softmax<<<BB * DD / 8, 256, 0, stream>>>(n2f, key16);
    query_softmax<<<ROWS / 8, 256, 0, stream>>>(n2f, q16);
    // contextT[b][e][d]: key[b] (384x1024) @ n1[b] (1024x384), swapped f16 out
    gemm_wmma<false, false, true, false, true>
        <<<dim3(DD / 128, DD / 128, BB), 256, 0, stream>>>(
        key16, n1_16, ctxT16, nullptr, nullptr, DD, DD, NTOK,
        (long long)DD * NTOK, (long long)NTOK * DD, (long long)DD * DD);
    // attended^T[b]: q16[b] (1024x384, K=d) x ctx (BT = ctxT[e][d]) -> f16
    gemm_wmma<false, false, true, true, false>
        <<<dim3(DD / 128, NTOK / 128, BB), 256, 0, stream>>>(
        q16, ctxT16, attd16, nullptr, nullptr, NTOK, DD, DD,
        (long long)NTOK * DD, (long long)DD * DD, (long long)NTOK * DD);
    // reprojection: (16384x384) x rp (BT=[768][384]) + rp_b -> rep f32
    gemm_wmma<true, false, false, true, false>
        <<<dim3(C2 / 128, ROWS / 128, 1), 256, 0, stream>>>(
        attd16, w16_rp, rep, rp_b, nullptr, ROWS, C2, DD, 0, 0, 0);
    // tx = concat(ch1, ch2) + LN_cn(rep)
    tx_fuse<<<ROWS, 256, 0, stream>>>(rep, cn_g, cn_b, ch1, ch2, tx);

    // ---- mix FFN ----
    ln_kernel<true, false><<<ROWS, 256, 0, stream>>>(tx, ln2_g, ln2_b, a16, nullptr, C2);
    gemm_wmma<true, false, false, true, false>
        <<<dim3(C4 / 128, ROWS / 128, 1), 256, 0, stream>>>(
        a16, w16_fc1T, h1, fc1_b, nullptr, ROWS, C4, C2, 0, 0, 0);
    dwconv_gelu<<<(unsigned)((long long)ROWS * C4 / 256), 256, 0, stream>>>(
        h1, dw_w, dw_b, a16);
    gemm_wmma<true, true, false, true, false>
        <<<dim3(C2 / 128, ROWS / 128, 1), 256, 0, stream>>>(
        a16, w16_fc2T, (float*)d_out, fc2_b, tx, ROWS, C2, C4, 0, 0, 0);
}
